// MultiHeadSelfAttention_77163382440439
// MI455X (gfx1250) — matrix operand order repaired and checked by reading, not compiled, not run on hardware
//
#include <hip/hip_runtime.h>
#include <hip/hip_bf16.h>

// ---------------------------------------------------------------------------
// MI455X (gfx1250) multi-head causal self-attention.
// bf16 WMMA GEMMs + flash attention with TDM (tensor_load_to_lds) K/V staging.
// B=4, S=2048, D=1024, H=16, DK=64.
// ---------------------------------------------------------------------------

typedef __attribute__((ext_vector_type(16))) __bf16 v16bf;
typedef __attribute__((ext_vector_type(8)))  __bf16 v8bf;
typedef __attribute__((ext_vector_type(4)))  __bf16 v4bf;
typedef __attribute__((ext_vector_type(8)))  float  v8f;
typedef __attribute__((ext_vector_type(4)))  float  v4f;

static constexpr int Bb = 4, S = 2048, D = 1024, H = 16, DK = 64;
static constexpr int TOK = Bb * S;     // 8192 tokens

#define WMMA_BF16(a, b, c) \
  __builtin_amdgcn_wmma_f32_16x16x32_bf16(false, (a), false, (b), (short)0, (c), false, false)

__device__ __forceinline__ __bf16 f2bf(float f) {
  union { float f; unsigned u; } x; x.f = f;
  unsigned r = x.u + 0x7FFFu + ((x.u >> 16) & 1u);   // round-to-nearest-even
  union { unsigned short s; __bf16 b; } o; o.s = (unsigned short)(r >> 16);
  return o.b;
}

__device__ __forceinline__ v16bf cat8(v8bf lo, v8bf hi) {
  return __builtin_shufflevector(lo, hi, 0,1,2,3,4,5,6,7,8,9,10,11,12,13,14,15);
}

// ---------------------------------------------------------------------------
// Tensor Data Mover: 2-D tile load Global -> LDS (CDNA5 ISA ch.8 D# layout).
// This toolchain exposes the 6-arg builtin:
//   (u32x4 g0, i32x8 g1, i32x4 g2, i32x4 g3, i32x8, i32 cpol)
// ---------------------------------------------------------------------------
#if __has_builtin(__builtin_amdgcn_tensor_load_to_lds)
#define HAVE_TDM 1
typedef __attribute__((ext_vector_type(4))) unsigned int u32x4;
typedef __attribute__((ext_vector_type(8))) int i32x8;
typedef __attribute__((ext_vector_type(4))) int i32x4;

__device__ __forceinline__ void tdm_load_2d(unsigned lds_off,
                                            unsigned long long gaddr,
                                            unsigned tdim0, unsigned tdim1,
                                            unsigned tile0, unsigned tile1,
                                            unsigned long long stride0) {
  u32x4 g0;
  g0[0] = 1u;                                        // count=1, user D#
  g0[1] = lds_off;                                   // lds_addr (bytes)
  g0[2] = (unsigned)(gaddr & 0xFFFFFFFFull);         // global_addr[31:0]
  g0[3] = (unsigned)((gaddr >> 32) & 0x01FFFFFFull)  // global_addr[56:32]
        | (2u << 30);                                // type=2 ("image")
  i32x8 g1;
  g1[0] = (int)(1u << 16);                           // data_size=2B, mask=0
  g1[1] = (int)((tdim0 & 0xFFFFu) << 16);            // tensor_dim0[15:0]
  g1[2] = (int)(((tdim0 >> 16) & 0xFFFFu) | ((tdim1 & 0xFFFFu) << 16));
  g1[3] = (int)(((tdim1 >> 16) & 0xFFFFu) | ((tile0 & 0xFFFFu) << 16));
  g1[4] = (int)(tile1 & 0xFFFFu);                    // tile_dim1; tile_dim2=0
  g1[5] = (int)(stride0 & 0xFFFFFFFFull);            // tensor_dim0_stride lo
  g1[6] = (int)((stride0 >> 32) & 0xFFFFull);        // stride0 hi; stride1=0
  g1[7] = 0;
  const i32x4 gz  = {0, 0, 0, 0};                    // 2-D: groups 2/3 unused
  const i32x8 gz8 = {0, 0, 0, 0, 0, 0, 0, 0};
  __builtin_amdgcn_tensor_load_to_lds(g0, g1, gz, gz, gz8, 0);
}
#endif

// ---------------------------------------------------------------------------
// One-time f32 -> bf16 conversion (4 elems / thread, b128 in, b64 out).
// ---------------------------------------------------------------------------
__global__ __launch_bounds__(256) void cvt_f32_bf16(const float* __restrict__ in,
                                                    __bf16* __restrict__ out,
                                                    int n4) {
  const int i = blockIdx.x * 256 + threadIdx.x;
  if (i < n4) {
    const v4f v = *(const v4f*)(in + 4 * (size_t)i);
    v4bf o;
#pragma unroll
    for (int j = 0; j < 4; ++j) o[j] = f2bf(v[j]);
    *(v4bf*)(out + 4 * (size_t)i) = o;
  }
}

// ---------------------------------------------------------------------------
// GEMM: Y = A @ W^T  (A: [8192 x 1024] bf16, W: [1024 x 1024] bf16 [out,in]).
// MODE 0: store bf16 [bh, s, dk]   (Q / K)
// MODE 1: store bf16 [bh, dk, s]   (V transposed)
// MODE 2: store f32  [t, n]        (final O projection)
// Block = 256 thr (8 waves). Wave tile = 32(M) x 64(N), K-step 32.
// ---------------------------------------------------------------------------
template <int MODE>
__global__ __launch_bounds__(256) void gemm_bf16(const __bf16* __restrict__ A,
                                                 const __bf16* __restrict__ Wb,
                                                 __bf16* __restrict__ obf,
                                                 float* __restrict__ of32) {
  const int lane = threadIdx.x & 31, w = threadIdx.x >> 5;
  const int g = lane >> 4, ln = lane & 15;
  const int mbase = blockIdx.x * 256 + w * 32;
  const int nbase = blockIdx.y * 64;

  const __bf16* Ap0 = A  + (size_t)(mbase + ln) * D + g * 8;
  const __bf16* Ap1 = A  + (size_t)(mbase + 16 + ln) * D + g * 8;
  const __bf16* Bp  = Wb + (size_t)(nbase + ln) * D + g * 8;

  v8f acc[2][4] = {};

  for (int kk = 0; kk < D; kk += 32) {
    const v16bf a0 = cat8(*(const v8bf*)(Ap0 + kk), *(const v8bf*)(Ap0 + kk + 16));
    const v16bf a1 = cat8(*(const v8bf*)(Ap1 + kk), *(const v8bf*)(Ap1 + kk + 16));

    __builtin_prefetch(Ap0 + kk + 32, 0, 0);
    __builtin_prefetch(Ap1 + kk + 32, 0, 0);
    __builtin_prefetch(Bp + kk + 32, 0, 0);

#pragma unroll
    for (int nt = 0; nt < 4; ++nt) {
      const __bf16* bp = Bp + (size_t)nt * 16 * D + kk;
      const v16bf b = cat8(*(const v8bf*)bp, *(const v8bf*)(bp + 16));
      acc[0][nt] = WMMA_BF16(a0, b, acc[0][nt]);
      acc[1][nt] = WMMA_BF16(a1, b, acc[1][nt]);
    }
  }

#pragma unroll
  for (int mt = 0; mt < 2; ++mt) {
#pragma unroll
    for (int nt = 0; nt < 4; ++nt) {
#pragma unroll
      for (int j = 0; j < 8; ++j) {
        const int t = mbase + mt * 16 + j + 8 * g;   // token
        const int n = nbase + nt * 16 + ln;          // out feature
        const float v = acc[mt][nt][j];
        if constexpr (MODE == 2) {
          of32[(size_t)t * D + n] = v;
        } else {
          const int b = t >> 11, s = t & (S - 1);
          const int h = n >> 6,  dk = n & (DK - 1);
          if constexpr (MODE == 0)
            obf[(((size_t)(b * H + h)) * S + s) * DK + dk] = f2bf(v);
          else
            obf[(((size_t)(b * H + h)) * DK + dk) * S + s] = f2bf(v);
        }
      }
    }
  }
}

// ---------------------------------------------------------------------------
// Flash attention, causal. Grid: (S/128, B*H). Block = 256 thr (8 waves).
// K/V tiles staged into LDS by the Tensor Data Mover (wave 0 issues,
// s_wait_tensorcnt + barrier publishes); all waves consume via ds_load.
// ---------------------------------------------------------------------------
__global__ __launch_bounds__(256) void flash_attn(const __bf16* __restrict__ Q,
                                                  const __bf16* __restrict__ K,
                                                  const __bf16* __restrict__ Vt,
                                                  __bf16* __restrict__ att) {
  __shared__ __bf16 Kl[32 * 64];        // 4 KB key tile, [key][d]
  __shared__ __bf16 Vl[64 * 32];        // 4 KB value tile, [d][key]
  __shared__ __bf16 pbuf[8][16 * 32];   // wave-private P tiles (8 KB)

  const int lane = threadIdx.x & 31, w = threadIdx.x >> 5;
  const int g = lane >> 4, ln = lane & 15;
  const int bh = blockIdx.y;
  const int q0 = blockIdx.x * 128 + w * 16;

  const __bf16* Qb = Q  + (size_t)bh * S * DK;
  const __bf16* Kb = K  + (size_t)bh * S * DK;
  const __bf16* Vb = Vt + (size_t)bh * DK * S;

  // Q fragments (16x32 each, covering d=[0,32) and [32,64))
  v16bf aQ0, aQ1;
  {
    const __bf16* p = Qb + (size_t)(q0 + ln) * DK + g * 8;
    aQ0 = cat8(*(const v8bf*)p,        *(const v8bf*)(p + 16));
    aQ1 = cat8(*(const v8bf*)(p + 32), *(const v8bf*)(p + 48));
  }

  v8f acc[4] = {};
  float mrow[8], lrow[8];
#pragma unroll
  for (int j = 0; j < 8; ++j) { mrow[j] = -__builtin_inff(); lrow[j] = 0.0f; }

  const float scale = 0.125f;               // DK^-0.5
  const int jend = q0 + 16;                 // this wave's causal key bound
  const int jblk_end = blockIdx.x * 128 + 128;  // block-shared key bound

  for (int j0 = 0; j0 < jblk_end; j0 += 32) {
    // ---- stage K (32x64, contiguous) and V (64x32, stride S) tiles
#if defined(HAVE_TDM)
    if (w == 0) {
      tdm_load_2d((unsigned)(size_t)(void*)Kl,
                  (unsigned long long)(size_t)(Kb + (size_t)j0 * DK),
                  /*tdim0=*/DK, /*tdim1=*/32, /*tile0=*/DK, /*tile1=*/32,
                  /*stride0=*/DK);
      tdm_load_2d((unsigned)(size_t)(void*)Vl,
                  (unsigned long long)(size_t)(Vb + j0),
                  /*tdim0=*/32, /*tdim1=*/DK, /*tile0=*/32, /*tile1=*/DK,
                  /*stride0=*/S);
      __builtin_amdgcn_s_wait_tensorcnt(0);
    }
#else
    {
      const int tid = threadIdx.x;
      *(v8bf*)&Kl[tid * 8] = *(const v8bf*)(Kb + (size_t)j0 * DK + tid * 8);
      const int dd = tid >> 2, cc = (tid & 3) * 8;
      *(v8bf*)&Vl[dd * 32 + cc] = *(const v8bf*)(Vb + (size_t)dd * S + j0 + cc);
    }
#endif
    __syncthreads();

    if (j0 < jend) {   // wave-uniform: EXEC all-ones inside
      // ---- scores: two 16x16 tiles (keys j0+[0,16) and j0+[16,32))
      v8f sc[2];
#pragma unroll
      for (int tt = 0; tt < 2; ++tt) {
        const __bf16* kp = &Kl[(tt * 16 + ln) * DK + g * 8];
        const v16bf b0 = cat8(*(const v8bf*)kp,        *(const v8bf*)(kp + 16));
        const v16bf b1 = cat8(*(const v8bf*)(kp + 32), *(const v8bf*)(kp + 48));
        v8f c = {};
        c = WMMA_BF16(aQ0, b0, c);
        c = WMMA_BF16(aQ1, b1, c);
        sc[tt] = c;
      }

      // ---- causal mask + online softmax
#pragma unroll
      for (int j = 0; j < 8; ++j) {
        const int r = q0 + j + 8 * g;
        float s0 = sc[0][j] * scale;
        float s1 = sc[1][j] * scale;
        if (j0 + ln > r)      s0 = -__builtin_inff();
        if (j0 + 16 + ln > r) s1 = -__builtin_inff();

        float mx = fmaxf(s0, s1);
#pragma unroll
        for (int msk = 1; msk < 16; msk <<= 1)
          mx = fmaxf(mx, __shfl_xor(mx, msk, 32));

        const float mnew  = fmaxf(mrow[j], mx);
        const float alpha = __expf(mrow[j] - mnew);
        const float p0 = __expf(s0 - mnew);
        const float p1 = __expf(s1 - mnew);
        float ps = p0 + p1;
#pragma unroll
        for (int msk = 1; msk < 16; msk <<= 1)
          ps += __shfl_xor(ps, msk, 32);

        lrow[j] = lrow[j] * alpha + ps;
        mrow[j] = mnew;
        acc[0][j] *= alpha; acc[1][j] *= alpha;
        acc[2][j] *= alpha; acc[3][j] *= alpha;

        pbuf[w][(j + 8 * g) * 32 + ln]      = f2bf(p0);
        pbuf[w][(j + 8 * g) * 32 + 16 + ln] = f2bf(p1);
      }

      // ---- P as A-fragment (16x32)
      const __bf16* pb = &pbuf[w][ln * 32 + g * 8];
      const v16bf aP = cat8(*(const v8bf*)pb, *(const v8bf*)(pb + 16));

      // ---- O += P @ V
#pragma unroll
      for (int nt = 0; nt < 4; ++nt) {
        const __bf16* vp = &Vl[(nt * 16 + ln) * 32 + g * 8];
        const v16bf bv = cat8(*(const v8bf*)vp, *(const v8bf*)(vp + 16));
        acc[nt] = WMMA_BF16(aP, bv, acc[nt]);
      }
    }
    __syncthreads();   // all waves done reading before next TDM overwrite
  }

  // ---- normalize and store bf16 to att[b, s, h*64 + d]
  const int b = bh >> 4, h = bh & (H - 1);
#pragma unroll
  for (int j = 0; j < 8; ++j) {
    const int r = q0 + j + 8 * g;
    const float inv = 1.0f / lrow[j];
    const size_t base = ((size_t)(b * S + r)) * D + h * DK;
#pragma unroll
    for (int nt = 0; nt < 4; ++nt)
      att[base + nt * 16 + ln] = f2bf(acc[nt][j] * inv);
  }
}

// ---------------------------------------------------------------------------
extern "C" void kernel_launch(void* const* d_in, const int* in_sizes, int n_in,
                              void* d_out, int out_size, void* d_ws, size_t ws_size,
                              hipStream_t stream) {
  const float* x  = (const float*)d_in[0];
  const float* Wq = (const float*)d_in[1];
  const float* Wk = (const float*)d_in[2];
  const float* Wv = (const float*)d_in[3];
  const float* Wo = (const float*)d_in[4];
  float* out = (float*)d_out;

  char* ws = (char*)d_ws;
  const size_t XSEG = (size_t)TOK * D * sizeof(__bf16);         // 16 MiB
  const size_t WSEG = (size_t)D * D * sizeof(__bf16);           //  2 MiB
  const size_t QSEG = (size_t)64 * S * DK * sizeof(__bf16);     // 16 MiB

  __bf16* xb   = (__bf16*)(ws);
  __bf16* Wqb  = (__bf16*)(ws + XSEG);
  __bf16* Wkb  = (__bf16*)(ws + XSEG + WSEG);
  __bf16* Wvb  = (__bf16*)(ws + XSEG + 2 * WSEG);
  __bf16* Wob  = (__bf16*)(ws + XSEG + 3 * WSEG);
  __bf16* Qbf  = (__bf16*)(ws + XSEG + 4 * WSEG);
  __bf16* Kbf  = (__bf16*)(ws + XSEG + 4 * WSEG + QSEG);
  __bf16* Vtbf = (__bf16*)(ws + XSEG + 4 * WSEG + 2 * QSEG);
  __bf16* attb = (__bf16*)(ws + XSEG + 4 * WSEG + 3 * QSEG);

  {
    const int nx4 = TOK * D / 4, nw4 = D * D / 4;
    cvt_f32_bf16<<<dim3((nx4 + 255) / 256), dim3(256), 0, stream>>>(x,  xb,  nx4);
    cvt_f32_bf16<<<dim3((nw4 + 255) / 256), dim3(256), 0, stream>>>(Wq, Wqb, nw4);
    cvt_f32_bf16<<<dim3((nw4 + 255) / 256), dim3(256), 0, stream>>>(Wk, Wkb, nw4);
    cvt_f32_bf16<<<dim3((nw4 + 255) / 256), dim3(256), 0, stream>>>(Wv, Wvb, nw4);
    cvt_f32_bf16<<<dim3((nw4 + 255) / 256), dim3(256), 0, stream>>>(Wo, Wob, nw4);
  }

  const dim3 gg(TOK / 256, D / 64), bb(256, 1, 1);
  gemm_bf16<0><<<gg, bb, 0, stream>>>(xb, Wqb, Qbf, nullptr);
  gemm_bf16<0><<<gg, bb, 0, stream>>>(xb, Wkb, Kbf, nullptr);
  gemm_bf16<1><<<gg, bb, 0, stream>>>(xb, Wvb, Vtbf, nullptr);

  flash_attn<<<dim3(S / 128, Bb * H), bb, 0, stream>>>(Qbf, Kbf, Vtbf, attb);

  gemm_bf16<2><<<gg, bb, 0, stream>>>(attb, Wob, nullptr, out);
}
